// BeeSenseSelector_91276644975184
// MI455X (gfx1250) — compile-verified, matrix operand order checked by reading, not run on hardware
//
#include <hip/hip_runtime.h>
#include <math.h>

// ---------------- problem constants ----------------
#define BATCH 32
#define CH    256
#define HW    12544              // 112*112 spatial positions
#define KTOP  128                // top-k = C/2
#define INV_HW (1.0f / 12544.0f)

#define POS_PER_BLOCK 128        // spatial positions per pooling block
#define CHUNKS_PER_B  (HW / POS_PER_BLOCK)   // 98

// apply kernel geometry: 1 block = 2048 contiguous float4 within one batch
#define F4_PER_BATCH  (HW * 64)               // 802816 float4 per batch
#define F4_PER_BLOCK  2048                    // 256 threads x 8 iters
#define BLOCKS_PER_B  (F4_PER_BATCH / F4_PER_BLOCK)   // 392

// ---------------- vector types ----------------
typedef __attribute__((ext_vector_type(16))) __bf16 v16bf;
typedef __attribute__((ext_vector_type(8)))  float  v8f;
typedef __attribute__((ext_vector_type(4)))  float  vf4;   // native clang vector
                                                           // (NT builtins accept it)

// ===================================================================
// 1) zero the pooled accumulator (atomics accumulate into it)
// ===================================================================
__global__ void bee_zero_kernel(float* __restrict__ p, int n) {
    int i = blockIdx.x * blockDim.x + threadIdx.x;
    if (i < n) p[i] = 0.0f;
}

// ===================================================================
// 2) global average pool:  pooled[b][c] = mean_s x[b][s][c]
//    256 threads cover 4 positions x 64 float4-channels per step,
//    b128 coalesced loads, LDS reduction, then 98 atomics/address.
// ===================================================================
__global__ void bee_pool_kernel(const float* __restrict__ x,
                                float* __restrict__ pooled) {
    __shared__ vf4 sm[256];
    const int t     = threadIdx.x;
    const int b     = blockIdx.x / CHUNKS_PER_B;
    const int chunk = blockIdx.x % CHUNKS_PER_B;
    const int posIn = t >> 6;      // 0..3 position within group of 4
    const int c4    = t & 63;      // float4 channel index

    const vf4* x4 = (const vf4*)x;
    vf4 s = {0.f, 0.f, 0.f, 0.f};
    const int base_pos = chunk * POS_PER_BLOCK;
    for (int g = 0; g < 32; ++g) {
        int pos = base_pos + g * 4 + posIn;
        s += x4[((size_t)b * HW + pos) * 64 + c4];   // global_load_b128
    }
    sm[t] = s;
    __syncthreads();
    if (t < 64) {
        vf4 r = (sm[t] + sm[t + 64]) + (sm[t + 128] + sm[t + 192]);
        float* dst = &pooled[b * CH + 4 * t];
        atomicAdd(dst + 0, r.x * INV_HW);
        atomicAdd(dst + 1, r.y * INV_HW);
        atomicAdd(dst + 2, r.z * INV_HW);
        atomicAdd(dst + 3, r.w * INV_HW);
    }
}

// ===================================================================
// 3) logits = pooled @ W + bias via v_wmma_f32_16x16x32_bf16.
//    NOTE: sigmoid is strictly monotonic and only the top-k RANK of
//    the scores is consumed downstream, so we rank raw logits and
//    skip the sigmoid entirely (kills the v_div/v_exp chains).
//    One block, 8 waves; 32 tiles of 16x16 (M=32, N=256), K=256 in
//    8 steps of 32. Layouts per ISA 7.12.2:
//      A: lane L -> row M=L%16; elem e -> Koff = (e<8 ? 8*(L/16)+e
//                                        : 16+8*(L/16)+(e-8))
//      B: mirrored with N = L%16 (column fetch of W).
//      C/D: VGPR r, lane L -> (M = r + 8*(L/16), N = L%16).
// ===================================================================
__global__ void bee_score_wmma_kernel(const float* __restrict__ pooled,
                                      const float* __restrict__ Wm,
                                      const float* __restrict__ bias,
                                      float* __restrict__ scores) {
    const int lane = threadIdx.x & 31;
    const int wave = threadIdx.x >> 5;   // 0..7
    const int hh   = lane >> 4;          // lane half: 0 or 1
    const int l16  = lane & 15;

    for (int tile = wave; tile < 32; tile += 8) {
        const int mt = tile >> 4;        // 0..1
        const int nt = tile & 15;        // 0..15
        const int m0 = mt * 16, n0 = nt * 16;
        const int mrow = m0 + l16;       // A row this lane supplies
        const int ncol = n0 + l16;       // B/D column this lane supplies
        const float* arow = pooled + mrow * CH;

        v8f acc = {0.f, 0.f, 0.f, 0.f, 0.f, 0.f, 0.f, 0.f};

        for (int k0 = 0; k0 < CH; k0 += 32) {
            // A: two contiguous 8-float groups per lane -> 4x b128 loads
            const float* ap = arow + k0 + 8 * hh;
            vf4 a0 = *(const vf4*)(ap + 0);
            vf4 a1 = *(const vf4*)(ap + 4);
            vf4 a2 = *(const vf4*)(ap + 16);
            vf4 a3 = *(const vf4*)(ap + 20);
            v16bf av;
            av[0]  = (__bf16)a0.x; av[1]  = (__bf16)a0.y;
            av[2]  = (__bf16)a0.z; av[3]  = (__bf16)a0.w;
            av[4]  = (__bf16)a1.x; av[5]  = (__bf16)a1.y;
            av[6]  = (__bf16)a1.z; av[7]  = (__bf16)a1.w;
            av[8]  = (__bf16)a2.x; av[9]  = (__bf16)a2.y;
            av[10] = (__bf16)a2.z; av[11] = (__bf16)a2.w;
            av[12] = (__bf16)a3.x; av[13] = (__bf16)a3.y;
            av[14] = (__bf16)a3.z; av[15] = (__bf16)a3.w;

            v16bf bv;
#pragma unroll
            for (int e = 0; e < 16; ++e) {
                int koff = (e < 8) ? (8 * hh + e) : (16 + 8 * hh + (e - 8));
                bv[e] = (__bf16)Wm[(k0 + koff) * CH + ncol];
            }
            acc = __builtin_amdgcn_wmma_f32_16x16x32_bf16(
                /*neg_a=*/false, av, /*neg_b=*/false, bv,
                /*c_mod=*/(short)0, acc, /*reuse_a=*/false, /*reuse_b=*/false);
        }

        const float bn = bias[ncol];
#pragma unroll
        for (int r = 0; r < 8; ++r) {
            int row = m0 + r + 8 * hh;
            scores[row * CH + ncol] = acc[r] + bn;   // raw logit; rank-equivalent
        }
    }
}

// ===================================================================
// 4) top-128 mask per row via rank counting (matches top_k tie rule:
//    equal values -> lower index wins).
// ===================================================================
__global__ void bee_mask_kernel(const float* __restrict__ scores,
                                float* __restrict__ mask) {
    __shared__ float s[CH];
    const int b = blockIdx.x;
    const int c = threadIdx.x;
    s[c] = scores[b * CH + c];
    __syncthreads();
    const float sc = s[c];
    int rank = 0;
#pragma unroll 8
    for (int j = 0; j < CH; ++j) {
        float sj = s[j];
        rank += (sj > sc) || (sj == sc && j < c);
    }
    mask[b * CH + c] = (rank < KTOP) ? 1.0f : 0.0f;
}

// ===================================================================
// 5) out = x * mask (broadcast over spatial). This 822 MB stream IS
//    the runtime (~35 us roofline at 23.3 TB/s). One block = 2048
//    contiguous float4 inside one batch: batch index from a cheap
//    constant division, mask float4 hoisted (loop-invariant per
//    thread), NT hints so the stream doesn't thrash L2.
// ===================================================================
__global__ void bee_apply_kernel(const float* __restrict__ x,
                                 const float* __restrict__ mask,
                                 float* __restrict__ out) {
    const vf4* x4 = (const vf4*)x;
    const vf4* m4 = (const vf4*)mask;
    vf4*       o4 = (vf4*)out;

    const int b   = blockIdx.x / BLOCKS_PER_B;   // const divisor -> mul
    const int blk = blockIdx.x % BLOCKS_PER_B;
    const size_t base = (size_t)b * F4_PER_BATCH + (size_t)blk * F4_PER_BLOCK;

    // base and the per-iter stride are multiples of 64, so the channel
    // group is loop-invariant: c4 = threadIdx.x & 63.
    const vf4 m = m4[b * 64 + (threadIdx.x & 63)];

#pragma unroll
    for (int it = 0; it < 8; ++it) {
        size_t i = base + (size_t)it * 256 + threadIdx.x;
        vf4 v = __builtin_nontemporal_load(&x4[i]);   // TH_NT b128
        v *= m;
        __builtin_nontemporal_store(v, &o4[i]);       // TH_NT b128
    }
}

// ===================================================================
extern "C" void kernel_launch(void* const* d_in, const int* in_sizes, int n_in,
                              void* d_out, int out_size, void* d_ws, size_t ws_size,
                              hipStream_t stream) {
    const float* x    = (const float*)d_in[0];   // [32,112,112,256] f32
    const float* Wm   = (const float*)d_in[1];   // [256,256] f32
    const float* bias = (const float*)d_in[2];   // [256] f32
    float*       out  = (float*)d_out;

    float* pooled = (float*)d_ws;                // 32*256 f32
    float* scores = pooled + BATCH * CH;         // 32*256 f32
    float* mask   = scores + BATCH * CH;         // 32*256 f32

    bee_zero_kernel<<<(BATCH * CH + 255) / 256, 256, 0, stream>>>(pooled, BATCH * CH);
    bee_pool_kernel<<<BATCH * CHUNKS_PER_B, 256, 0, stream>>>(x, pooled);
    bee_score_wmma_kernel<<<1, 256, 0, stream>>>(pooled, Wm, bias, scores);
    bee_mask_kernel<<<BATCH, 256, 0, stream>>>(scores, mask);
    bee_apply_kernel<<<BATCH * BLOCKS_PER_B, 256, 0, stream>>>(x, mask, out);
}
